// factorMLPFunction_57896159150712
// MI455X (gfx1250) — compile-verified, matrix operand order checked by reading, not run on hardware
//
#include <hip/hip_runtime.h>

// Problem sizes (fixed by the reference): B=1024, N=512, F=128, H=512.
#define BB 1024
#define NN 512
#define FF 128
#define HH 512

typedef __attribute__((ext_vector_type(16))) __bf16 v16bf;
typedef __attribute__((ext_vector_type(8)))  float  v8f;

union Frag16 { uint4 u[2]; v16bf v; };

#if defined(__gfx1250__) && __has_builtin(__builtin_amdgcn_global_load_async_to_lds_b128)
#define HAVE_ASYNC_LDS 1
#else
#define HAVE_ASYNC_LDS 0
#endif

#if HAVE_ASYNC_LDS
static __device__ __forceinline__ void async_ld_b128(const unsigned short* g,
                                                     unsigned short* l) {
    typedef int v4i_vs __attribute__((vector_size(16)));
    __builtin_amdgcn_global_load_async_to_lds_b128(
        (__attribute__((address_space(1))) v4i_vs*)(g),
        (__attribute__((address_space(3))) v4i_vs*)(l), 0, 0);
}
#if __has_builtin(__builtin_amdgcn_s_wait_asynccnt)
#define WAIT_ASYNC(n) __builtin_amdgcn_s_wait_asynccnt(n)
#else
#define WAIT_ASYNC(n) asm volatile("s_wait_asynccnt %0" ::"i"(n) : "memory")
#endif
#endif

// CDNA5 has a native V_TANH_F32 transcendental (single TRANS op, co-executes
// with XDL/WMMA) -- use it instead of the ~20-op ocml expansion if exposed.
#if __has_builtin(__builtin_amdgcn_tanhf)
#define TANH(x) __builtin_amdgcn_tanhf(x)
#else
#define TANH(x) tanhf(x)
#endif

static __device__ __forceinline__ unsigned short f2bf(float f) {
    unsigned int u = __float_as_uint(f);
    unsigned int r = (u + 0x7FFFu + ((u >> 16) & 1u)) >> 16;   // round-to-nearest-even
    return (unsigned short)r;
}

// ---------------------------------------------------------------------------
// Pack a weight tensor W (outer, H=512 rows, Kdim cols) fp32 into bf16 WMMA
// B-fragment order: dst[tile=((o*KB+kb)*32+hb)][lane][i], 32B contiguous per
// lane.  16-bit operand lane/K map (ISA 7.12.2): lanes 0-15 -> K {0..7,16..23},
// lanes 16-31 -> K {8..15,24..31}; column index = lane%16.
// ---------------------------------------------------------------------------
__global__ __launch_bounds__(256) void pack_weights(
    const float* __restrict__ W, unsigned short* __restrict__ Frag, int Kdim)
{
    const int KB   = Kdim >> 5;
    const int t    = threadIdx.x;
    const int tile = blockIdx.x * 8 + (t >> 5);
    const int lane = t & 31, g = lane >> 4, lr = lane & 15;
    const int hb = tile & 31;
    const int kb = (tile >> 5) % KB;
    const int o  = tile / (KB * 32);

    const float* src = W + ((size_t)o * HH + hb * 16 + lr) * Kdim + kb * 32 + g * 8;
    float4 r0a = *(const float4*)(src);
    float4 r0b = *(const float4*)(src + 4);
    float4 r1a = *(const float4*)(src + 16);
    float4 r1b = *(const float4*)(src + 20);

    union { unsigned short h[16]; uint4 u[2]; } pk;
    pk.h[0]  = f2bf(r0a.x); pk.h[1]  = f2bf(r0a.y); pk.h[2]  = f2bf(r0a.z); pk.h[3]  = f2bf(r0a.w);
    pk.h[4]  = f2bf(r0b.x); pk.h[5]  = f2bf(r0b.y); pk.h[6]  = f2bf(r0b.z); pk.h[7]  = f2bf(r0b.w);
    pk.h[8]  = f2bf(r1a.x); pk.h[9]  = f2bf(r1a.y); pk.h[10] = f2bf(r1a.z); pk.h[11] = f2bf(r1a.w);
    pk.h[12] = f2bf(r1b.x); pk.h[13] = f2bf(r1b.y); pk.h[14] = f2bf(r1b.z); pk.h[15] = f2bf(r1b.w);

    uint4* d = (uint4*)(Frag + (size_t)tile * 512 + lane * 16);
    d[0] = pk.u[0];
    d[1] = pk.u[1];
}

// ---------------------------------------------------------------------------
// Um[f][b][n] = bf16( mask_n2m[b][n][f] * x[b][n] )   -- LDS tile transpose,
// coalesced read along f, coalesced write along n.
// ---------------------------------------------------------------------------
__global__ __launch_bounds__(256) void prep_um(
    const float* __restrict__ mask, const float* __restrict__ x,
    unsigned short* __restrict__ Um)
{
    const int t  = threadIdx.x;
    const int b  = blockIdx.y;
    const int n0 = blockIdx.x * 32;
    __shared__ float tile[32 * 132];
    __shared__ float sx[32];
    if (t < 32) sx[t] = x[(size_t)b * NN + n0 + t];
#pragma unroll
    for (int it = 0; it < 16; ++it) {
        int lin = t + 256 * it;
        int f = lin & 127, nl = lin >> 7;
        tile[nl * 132 + f] = mask[((size_t)b * NN + n0 + nl) * FF + f];
    }
    __syncthreads();
#pragma unroll
    for (int it = 0; it < 16; ++it) {
        int lin = t + 256 * it;
        int col = lin & 31, f = lin >> 5;
        Um[((size_t)f * BB + b) * NN + n0 + col] = f2bf(tile[col * 132 + f] * sx[col]);
    }
}

// ---------------------------------------------------------------------------
// Vm[n][b][ff] = bf16( mask_m2n_T[b][ff][n] * g_x[b][ff] )
// ---------------------------------------------------------------------------
__global__ __launch_bounds__(256) void prep_vm(
    const float* __restrict__ mask, const float* __restrict__ gx,
    unsigned short* __restrict__ Vm)
{
    const int t   = threadIdx.x;
    const int b   = blockIdx.z;
    const int ff0 = blockIdx.y * 32;
    const int n0  = blockIdx.x * 32;
    __shared__ float tile[32 * 33];
    __shared__ float sg[32];
    if (t < 32) sg[t] = gx[(size_t)b * FF + ff0 + t];
#pragma unroll
    for (int it = 0; it < 4; ++it) {
        int lin = t + 256 * it;
        int nl = lin & 31, fl = lin >> 5;
        tile[fl * 33 + nl] = mask[((size_t)b * FF + ff0 + fl) * NN + n0 + nl];
    }
    __syncthreads();
#pragma unroll
    for (int it = 0; it < 4; ++it) {
        int lin = t + 256 * it;
        int fl = lin & 31, nl = lin >> 5;
        Vm[((size_t)(n0 + nl) * BB + b) * FF + ff0 + fl] = f2bf(tile[fl * 33 + nl] * sg[fl]);
    }
}

// ---------------------------------------------------------------------------
// Fused grouped GEMM + tanh + second-layer dot + tanh.
//   A    : (NFgroups, B, Kdim) bf16   (pre-masked inputs)
//   Bfrag: (NFgroups, KB, 32hb, 32lane, 16) bf16 fragment order
//   W2   : (NFgroups, 512) fp32
//   out[(b)*outStride + group] = tanh( sum_h tanh((A·W1ᵀ)[b,h]) * W2[h] )
// Grid: (B/128, NFgroups), block 256 = 8 waves; each wave owns a 16-row
// M-block, 16 v8f accumulators per half-H pass, K tiled via double-buffered
// LDS slabs filled by GLOBAL_LOAD_ASYNC_TO_LDS (ASYNCcnt) when available.
// B fragments are double-buffered in registers so ds_load of tile hb+1
// overlaps the WMMA of tile hb (dscnt<=1 instead of 0).
// ---------------------------------------------------------------------------
__global__ __launch_bounds__(256) void gemm_fused(
    const unsigned short* __restrict__ A,
    const unsigned short* __restrict__ Bfrag,
    const float* __restrict__ W2,
    float* __restrict__ out,
    int Kdim, int outStride)
{
    const int t   = threadIdx.x;
    const int fid = blockIdx.y;
    const int b0  = blockIdx.x * 128;
    const int KB  = Kdim >> 5;
    const int wave = t >> 5, lane = t & 31, g = lane >> 4, lr = lane & 15;
    const int m0 = wave * 16;

    __shared__ float sw2[512];
    sw2[t]       = W2[(size_t)fid * HH + t];
    sw2[t + 256] = W2[(size_t)fid * HH + t + 256];

    float gp[8];
#pragma unroll
    for (int r = 0; r < 8; ++r) gp[r] = 0.0f;

    const size_t Abase = ((size_t)fid * BB + b0) * Kdim;
    const size_t Fbase = (size_t)fid * KB * 16384;
    const int s0 = g * 8;

#if HAVE_ASYNC_LDS
    __shared__ unsigned short sA[2][128 * 40];   // padded rows, 16B-aligned
    __shared__ unsigned short sB[2][16 * 512];   // 16 B-fragment tiles / half
    const int arow = t >> 1;
    const int ac0  = (t & 1) * 16;

    for (int p = 0; p < 2; ++p) {
        v8f acc[16];
#pragma unroll
        for (int i = 0; i < 16; ++i) acc[i] = (v8f){0, 0, 0, 0, 0, 0, 0, 0};

        // 6 async b128 per thread per slab: 2 for A (32B), 4 for B (64B)
        auto issue = [&](int kb, int buf) {
            const unsigned short* ga = A + Abase + (size_t)arow * Kdim + kb * 32 + ac0;
            unsigned short* la = &sA[buf][arow * 40 + ac0];
            async_ld_b128(ga, la);
            async_ld_b128(ga + 8, la + 8);
            const unsigned short* gb =
                Bfrag + Fbase + (size_t)kb * 16384 + p * 8192 + t * 32;
            unsigned short* lb = &sB[buf][t * 32];
#pragma unroll
            for (int j = 0; j < 4; ++j) async_ld_b128(gb + j * 8, lb + j * 8);
        };

        issue(0, 0);
        for (int kb = 0; kb < KB; ++kb) {
            const int cur = kb & 1;
            if (kb + 1 < KB) {               // prefetch next slab into other buffer
                issue(kb + 1, cur ^ 1);
                WAIT_ASYNC(6);               // in-order: current slab's 6 are done
            } else {
                WAIT_ASYNC(0);
            }
            __syncthreads();                 // all waves' DMA visible

            Frag16 af;
            af.u[0] = *(const uint4*)&sA[cur][(m0 + lr) * 40 + s0];
            af.u[1] = *(const uint4*)&sA[cur][(m0 + lr) * 40 + s0 + 16];

            Frag16 bf[2];                    // register double-buffer for B frags
            {
                const uint4* lb0 = (const uint4*)&sB[cur][lane * 16];
                bf[0].u[0] = lb0[0]; bf[0].u[1] = lb0[1];
            }
#pragma unroll
            for (int hb = 0; hb < 16; ++hb) {
                if (hb + 1 < 16) {
                    const uint4* lbn =
                        (const uint4*)&sB[cur][(hb + 1) * 512 + lane * 16];
                    bf[(hb + 1) & 1].u[0] = lbn[0];
                    bf[(hb + 1) & 1].u[1] = lbn[1];
                }
                acc[hb] = __builtin_amdgcn_wmma_f32_16x16x32_bf16(
                    false, af.v, false, bf[hb & 1].v, (short)0, acc[hb], false, false);
            }
            __syncthreads();                 // readers done before buffer reuse
        }

#pragma unroll
        for (int hb = 0; hb < 16; ++hb) {
            float w = sw2[(p * 16 + hb) * 16 + lr];
#pragma unroll
            for (int r = 0; r < 8; ++r)
                gp[r] += TANH(acc[hb][r]) * w;
        }
    }
#else
    __shared__ unsigned short sA1[128 * 40];
    __shared__ unsigned short sB1[16 * 512];
    for (int p = 0; p < 2; ++p) {
        v8f acc[16];
#pragma unroll
        for (int i = 0; i < 16; ++i) acc[i] = (v8f){0, 0, 0, 0, 0, 0, 0, 0};
        for (int kb = 0; kb < KB; ++kb) {
            __syncthreads();
            {
                int row = t >> 1, c0 = (t & 1) * 16;
                const uint4* gsa = (const uint4*)(A + Abase + (size_t)row * Kdim + kb * 32 + c0);
                uint4 a0 = gsa[0], a1 = gsa[1];
                uint4* lda = (uint4*)&sA1[row * 40 + c0];
                lda[0] = a0; lda[1] = a1;
            }
            {
                const uint4* gsb = (const uint4*)(Bfrag + Fbase + (size_t)kb * 16384 + p * 8192);
                uint4* ldb = (uint4*)sB1;
#pragma unroll
                for (int j = 0; j < 4; ++j) ldb[t * 4 + j] = gsb[t * 4 + j];
            }
            __syncthreads();
            Frag16 af;
            af.u[0] = *(const uint4*)&sA1[(m0 + lr) * 40 + s0];
            af.u[1] = *(const uint4*)&sA1[(m0 + lr) * 40 + s0 + 16];
            Frag16 bf[2];
            {
                const uint4* lb0 = (const uint4*)&sB1[lane * 16];
                bf[0].u[0] = lb0[0]; bf[0].u[1] = lb0[1];
            }
#pragma unroll
            for (int hb = 0; hb < 16; ++hb) {
                if (hb + 1 < 16) {
                    const uint4* lbn = (const uint4*)&sB1[(hb + 1) * 512 + lane * 16];
                    bf[(hb + 1) & 1].u[0] = lbn[0];
                    bf[(hb + 1) & 1].u[1] = lbn[1];
                }
                acc[hb] = __builtin_amdgcn_wmma_f32_16x16x32_bf16(
                    false, af.v, false, bf[hb & 1].v, (short)0, acc[hb], false, false);
            }
        }
#pragma unroll
        for (int hb = 0; hb < 16; ++hb) {
            float w = sw2[(p * 16 + hb) * 16 + lr];
#pragma unroll
            for (int r = 0; r < 8; ++r)
                gp[r] += TANH(acc[hb][r]) * w;
        }
    }
#endif

    // reduce over the 16 lanes of each half-wave (columns), write tanh(sum)
#pragma unroll
    for (int r = 0; r < 8; ++r) {
        float s = gp[r];
        s += __shfl_xor(s, 1);
        s += __shfl_xor(s, 2);
        s += __shfl_xor(s, 4);
        s += __shfl_xor(s, 8);
        if (lr == 0)
            out[(size_t)(b0 + m0 + g * 8 + r) * outStride + fid] = TANH(s);
    }
}

// ---------------------------------------------------------------------------
extern "C" void kernel_launch(void* const* d_in, const int* in_sizes, int n_in,
                              void* d_out, int out_size, void* d_ws, size_t ws_size,
                              hipStream_t stream) {
    (void)in_sizes; (void)n_in; (void)out_size; (void)ws_size;
    const float* x     = (const float*)d_in[0];
    const float* m_n2m = (const float*)d_in[1];   // (B,N,F)
    const float* m_m2n = (const float*)d_in[2];   // (B,F,N)
    const float* Wfac1 = (const float*)d_in[3];   // (F,H,N)
    const float* Wfac2 = (const float*)d_in[4];   // (F,H)
    const float* Wvar1 = (const float*)d_in[5];   // (N,H,F)
    const float* Wvar2 = (const float*)d_in[6];   // (N,H)
    float* out = (float*)d_out;

    char* ws = (char*)d_ws;
    const size_t SZ_WFAC = (size_t)FF * HH * NN * 2;   // 64 MiB bf16 frags
    const size_t SZ_WVAR = (size_t)NN * HH * FF * 2;   // 64 MiB
    const size_t SZ_UMVM = (size_t)FF * BB * NN * 2;   // 128 MiB (Um aliases Vm)
    unsigned short* wfacF = (unsigned short*)(ws);
    unsigned short* wvarF = (unsigned short*)(ws + SZ_WFAC);
    unsigned short* UmVm  = (unsigned short*)(ws + SZ_WFAC + SZ_WVAR);
    float*          gx    = (float*)(ws + SZ_WFAC + SZ_WVAR + SZ_UMVM);

    // 1) pack weights to bf16 WMMA-fragment order
    pack_weights<<<dim3((FF * 16 * 32) / 8), 256, 0, stream>>>(Wfac1, wfacF, NN);
    pack_weights<<<dim3((NN * 4 * 32) / 8), 256, 0, stream>>>(Wvar1, wvarF, FF);
    // 2) stage A input: Um[f][b][n] = mask*x (bf16)
    prep_um<<<dim3(NN / 32, BB), 256, 0, stream>>>(m_n2m, x, UmVm);
    // 3) stage A fused GEMM -> g_x (B,F)
    gemm_fused<<<dim3(BB / 128, FF), 256, 0, stream>>>(UmVm, wfacF, Wfac2, gx, NN, FF);
    // 4) stage B input: Vm[n][b][ff] = mask*g_x (bf16), reuses Um buffer
    prep_vm<<<dim3(NN / 32, FF / 32, BB), 256, 0, stream>>>(m_m2n, gx, UmVm);
    // 5) stage B fused GEMM -> f_x (B,N)
    gemm_fused<<<dim3(BB / 128, NN), 256, 0, stream>>>(UmVm, wvarF, Wvar2, out, FF, NN);
}
